// GeneralizedPooling_42185168781653
// MI455X (gfx1250) — matrix operand order repaired and checked by reading, not compiled, not run on hardware
//
#include <hip/hip_runtime.h>
#include <hip/hip_bf16.h>
#include <math.h>

typedef __bf16 bf16_t;
typedef __attribute__((ext_vector_type(16))) __bf16 v16bf;
typedef __attribute__((ext_vector_type(8)))  __bf16 v8bf;
typedef __attribute__((ext_vector_type(8)))  float  v8f;

// Problem constants
constexpr int B_ = 16, S_ = 4096, E_ = 768, H_ = 4, A_ = 128;
constexpr float ALPHA_ = 0.2f;
constexpr int NSPLIT = 16;              // S-splits for online-softmax partials
constexpr int SCHUNK = S_ / NSPLIT;     // 256 rows per split

// Workspace layout (bytes)
constexpr size_t OFF_W1T  = 0;                                   // [H][A][E] bf16 (B-matrix N-major)
constexpr size_t SZ_W1T   = (size_t)H_ * A_ * E_ * 2;
constexpr size_t OFF_W2T  = OFF_W1T + SZ_W1T;                    // [H][E][A] bf16 (B-matrix N-major)
constexpr size_t SZ_W2T   = (size_t)H_ * E_ * A_ * 2;
constexpr size_t OFF_HACT = OFF_W2T + SZ_W2T;                    // [B][H][S][A] bf16
constexpr size_t SZ_HACT  = (size_t)B_ * H_ * S_ * A_ * 2;
constexpr size_t OFF_PART = OFF_HACT + SZ_HACT;                  // [B][H][NSPLIT][3][E] f32

// ---------------------------------------------------------------------------
// K0: convert W1 -> W1t[h][a][e] bf16, W2 -> W2t[h][e][a] bf16 (transposed so
// B-matrix fragments (col N fixed, K contiguous) are contiguous 32B loads).
// ---------------------------------------------------------------------------
__global__ void k_convert_weights(const float* __restrict__ W1,
                                  const float* __restrict__ W2,
                                  bf16_t* __restrict__ W1t,
                                  bf16_t* __restrict__ W2t) {
    int i = blockIdx.x * blockDim.x + threadIdx.x;
    const int n = H_ * E_ * A_;
    if (i >= n) return;
    int h   = i / (E_ * A_);
    int rem = i % (E_ * A_);
    int e   = rem / A_;
    int a   = rem % A_;
    W1t[((size_t)h * A_ + a) * E_ + e] = (bf16_t)W1[((size_t)h * E_ + e) * A_ + a];
    W2t[((size_t)h * E_ + e) * A_ + a] = (bf16_t)W2[((size_t)h * A_ + a) * E_ + e];
}

// ---------------------------------------------------------------------------
// K1: Hact[b][h][s][a] = leaky(x[b] @ W1[h] + b1[h])   via v_wmma_f32_16x16x32_bf16
// One wave: 16 s-rows x 128 a-cols (8 WMMA tiles), K-loop over E in steps of 32.
// ---------------------------------------------------------------------------
__global__ void __launch_bounds__(128) k_gemm1(const float* __restrict__ x,
                                               const bf16_t* __restrict__ W1t,
                                               const float* __restrict__ b1,
                                               bf16_t* __restrict__ hact) {
    const int lane = threadIdx.x & 31;
    const int wave = threadIdx.x >> 5;
    const int half = lane >> 4;
    const int l16  = lane & 15;

    const int ST    = S_ / 64;          // 64 s-rows per workgroup (4 waves)
    const int stile = blockIdx.x % ST;
    const int bh    = blockIdx.x / ST;
    const int h     = bh % H_;
    const int b     = bh / H_;
    const int s0    = stile * 64 + wave * 16;

    v8f acc[8] = {};

    // A-fragment row for this lane (f32 source, converted to bf16 on the fly)
    const float* xrow = x + ((size_t)b * S_ + s0 + l16) * E_;

#pragma unroll 1
    for (int kk = 0; kk < E_; kk += 32) {
        // A 16x32 bf16 layout: elems 0..7 -> K = kk + half*16 + (0..7)
        //                      elems 8..15 -> K = kk + half*16 + 8 + (0..7)
        v8f lo = *(const v8f*)(xrow + kk + half * 16);
        v8f hi = *(const v8f*)(xrow + kk + half * 16 + 8);
        v16bf af;
#pragma unroll
        for (int i = 0; i < 8; ++i) {
            af[i]     = (bf16_t)lo[i];
            af[8 + i] = (bf16_t)hi[i];
        }
#pragma unroll
        for (int t = 0; t < 8; ++t) {
            // B 32x16 bf16: lane = col n, elems 0..15 -> K = kk + half*16 + (0..15)
            const bf16_t* bp = W1t + ((size_t)h * A_ + t * 16 + l16) * E_ + kk + half * 16;
            v16bf bfrag = *(const v16bf*)bp;
            acc[t] = __builtin_amdgcn_wmma_f32_16x16x32_bf16(
                false, af, false, bfrag, (short)0, acc[t], false, false);
        }
    }

    // Epilogue: bias + leaky relu, store bf16
#pragma unroll
    for (int t = 0; t < 8; ++t) {
        const int a    = t * 16 + l16;
        const float bi = b1[h * A_ + a];
#pragma unroll
        for (int r = 0; r < 8; ++r) {
            float v = acc[t][r] + bi;
            v = (v >= 0.0f) ? v : ALPHA_ * v;
            const int row = s0 + r + half * 8;   // C/D layout: M = r + half*8
            hact[(((size_t)b * H_ + h) * S_ + row) * A_ + a] = (bf16_t)v;
        }
    }
}

// ---------------------------------------------------------------------------
// K2: logits = Hact @ W2[h] + b2[h], fused online softmax over s within an
// S-split. Each workgroup: (b, h, split of 256 rows); each wave owns 192
// e-columns (12 WMMA n-tiles) and keeps (max, sum_exp, sum_x*exp) in registers.
// ---------------------------------------------------------------------------
__global__ void __launch_bounds__(128) k_gemm2_softmax(const float* __restrict__ x,
                                                       const bf16_t* __restrict__ hact,
                                                       const bf16_t* __restrict__ W2t,
                                                       const float* __restrict__ b2,
                                                       float* __restrict__ part) {
    const int lane = threadIdx.x & 31;
    const int wave = threadIdx.x >> 5;
    const int half = lane >> 4;
    const int l16  = lane & 15;

    const int split = blockIdx.x % NSPLIT;
    const int bh    = blockIdx.x / NSPLIT;
    const int h     = bh % H_;
    const int b     = bh / H_;

    constexpr int NT = (E_ / 4) / 16;   // 12 n-tiles per wave
    const int nbase0 = wave * (E_ / 4);

    float m_st[NT], l_st[NT], t_st[NT];
#pragma unroll
    for (int t = 0; t < NT; ++t) {
        m_st[t] = -__builtin_inff();
        l_st[t] = 0.0f;
        t_st[t] = 0.0f;
    }

    const size_t hbase = ((size_t)b * H_ + h) * S_;

#pragma unroll 1
    for (int st = 0; st < SCHUNK / 16; ++st) {
        const int s0 = split * SCHUNK + st * 16;

        // Preload all 4 A-fragments of the Hact 16x128 tile (bf16 source)
        v16bf af[4];
        const bf16_t* hrow = hact + (hbase + s0 + l16) * A_;
#pragma unroll
        for (int kkI = 0; kkI < 4; ++kkI) {
            const int kk = kkI * 32;
            v8bf lo = *(const v8bf*)(hrow + kk + half * 16);
            v8bf hi = *(const v8bf*)(hrow + kk + half * 16 + 8);
#pragma unroll
            for (int i = 0; i < 8; ++i) {
                af[kkI][i]     = lo[i];
                af[kkI][8 + i] = hi[i];
            }
        }

#pragma unroll
        for (int t = 0; t < NT; ++t) {
            const int e = nbase0 + t * 16 + l16;   // this lane's e-column
            v8f acc = {};
            const bf16_t* wrow = W2t + ((size_t)h * E_ + e) * A_;
#pragma unroll
            for (int kkI = 0; kkI < 4; ++kkI) {
                v16bf bfrag = *(const v16bf*)(wrow + kkI * 32 + half * 16);
                acc = __builtin_amdgcn_wmma_f32_16x16x32_bf16(
                    false, af[kkI], false, bfrag, (short)0, acc, false, false);
            }

            const float bias = b2[h * E_ + e];
            float lg[8];
            float tm = -__builtin_inff();
#pragma unroll
            for (int r = 0; r < 8; ++r) {
                lg[r] = acc[r] + bias;
                tm = fmaxf(tm, lg[r]);
            }
            // combine the two half-wave row groups (rows r vs r+8), same column
            tm = fmaxf(tm, __shfl_xor(tm, 16, 32));

            const float m_new = fmaxf(m_st[t], tm);
            const float scale = __expf(m_st[t] - m_new);   // 0 on first tile (-inf)

            float se = 0.0f, sx = 0.0f;
#pragma unroll
            for (int r = 0; r < 8; ++r) {
                const int row = s0 + r + half * 8;
                const float p = __expf(lg[r] - m_new);
                se += p;
                sx += x[((size_t)b * S_ + row) * E_ + e] * p;
            }
            se += __shfl_xor(se, 16, 32);
            sx += __shfl_xor(sx, 16, 32);

            l_st[t] = l_st[t] * scale + se;
            t_st[t] = t_st[t] * scale + sx;
            m_st[t] = m_new;
        }
    }

    // Write per-split partials (state is replicated in both half-waves)
    if (lane < 16) {
        const size_t base = (((size_t)b * H_ + h) * NSPLIT + split) * 3 * (size_t)E_;
#pragma unroll
        for (int t = 0; t < NT; ++t) {
            const int e = nbase0 + t * 16 + l16;
            part[base + 0 * (size_t)E_ + e] = m_st[t];
            part[base + 1 * (size_t)E_ + e] = l_st[t];
            part[base + 2 * (size_t)E_ + e] = t_st[t];
        }
    }
}

// ---------------------------------------------------------------------------
// K3: merge NSPLIT online-softmax partials -> out[b][h][e] = sum(x*w)/sum(w)
// ---------------------------------------------------------------------------
__global__ void k_reduce(const float* __restrict__ part, float* __restrict__ out) {
    const int i = blockIdx.x * blockDim.x + threadIdx.x;
    if (i >= B_ * H_ * E_) return;
    const int e  = i % E_;
    const int bh = i / E_;
    const size_t base0 = (size_t)bh * NSPLIT * 3 * E_;

    float m = -__builtin_inff();
    for (int s = 0; s < NSPLIT; ++s)
        m = fmaxf(m, part[base0 + (size_t)s * 3 * E_ + e]);

    float l = 0.0f, t = 0.0f;
    for (int s = 0; s < NSPLIT; ++s) {
        const size_t pb = base0 + (size_t)s * 3 * E_;
        const float sc = __expf(part[pb + e] - m);
        l += part[pb + (size_t)E_ + e] * sc;
        t += part[pb + 2 * (size_t)E_ + e] * sc;
    }
    out[i] = t / l;
}

// ---------------------------------------------------------------------------
extern "C" void kernel_launch(void* const* d_in, const int* in_sizes, int n_in,
                              void* d_out, int out_size, void* d_ws, size_t ws_size,
                              hipStream_t stream) {
    const float* x  = (const float*)d_in[0];
    const float* W1 = (const float*)d_in[1];
    const float* b1 = (const float*)d_in[2];
    const float* W2 = (const float*)d_in[3];
    const float* b2 = (const float*)d_in[4];
    float* out = (float*)d_out;

    char* ws = (char*)d_ws;
    bf16_t* W1t  = (bf16_t*)(ws + OFF_W1T);
    bf16_t* W2t  = (bf16_t*)(ws + OFF_W2T);
    bf16_t* hact = (bf16_t*)(ws + OFF_HACT);
    float*  part = (float*)(ws + OFF_PART);

    const int nconv = H_ * E_ * A_;
    k_convert_weights<<<(nconv + 255) / 256, 256, 0, stream>>>(W1, W2, W1t, W2t);

    k_gemm1<<<B_ * H_ * (S_ / 64), 128, 0, stream>>>(x, W1t, b1, hact);

    k_gemm2_softmax<<<B_ * H_ * NSPLIT, 128, 0, stream>>>(x, hact, W2t, b2, part);

    const int nred = B_ * H_ * E_;
    k_reduce<<<(nred + 255) / 256, 256, 0, stream>>>(part, out);
}
